// PointNet2Seg_8813272891485
// MI455X (gfx1250) — compile-verified, matrix operand order checked by reading, not run on hardware
//
#include <hip/hip_runtime.h>

// ---------------------------------------------------------------------------
// PointNet++ segmentation forward for MI455X (gfx1250, wave32).
// Pointwise-conv GEMMs run on v_wmma_f32_16x16x32_bf16 with f32 accumulate.
// Both GEMM operands are stored in WMMA-fragment-tiled order so every
// fragment load is a pair of aligned b128 loads (no scalar u16 gathering).
// ---------------------------------------------------------------------------

typedef __attribute__((ext_vector_type(16))) __bf16          v16bf;
typedef __attribute__((ext_vector_type(8)))  float           v8f;
typedef __attribute__((ext_vector_type(8)))  unsigned short  ushort8;

union FragBF {
    v16bf          v;
    ushort8        h[2];
    unsigned short u[16];
};

__device__ __forceinline__ unsigned short f2bf(float x) {
    unsigned int u = __float_as_uint(x);
    u = (u + 0x7fffu + ((u >> 16) & 1u)) >> 16;   // round-to-nearest-even
    return (unsigned short)u;
}

// Tiled B-operand index: element (channel c, column n) of a [Cp, N] matrix,
// stored as 512-element 32x16 tiles in the exact v_wmma B-fragment layout:
//   k = k0 + (e>=8 ? 16 : 0) + hi*8 + (e&7),  lane = hi*16 + (n&15)
__device__ __forceinline__ size_t bidx(int c, int n, int N) {
    int kt = c >> 5, kr = c & 31;
    int hi = (kr >> 3) & 1;
    int e  = ((kr >> 4) & 1) * 8 + (kr & 7);
    return ((size_t)kt * (N >> 4) + (n >> 4)) * 512
         + (size_t)((hi << 4) | (n & 15)) * 16 + e;
}

// ---------------------------------------------------------------------------
// WMMA GEMM:  Y[O,N] = A_bf16[O,Cp] * X_bf16[Cp,N] + bias[O]
// A and X pre-tiled in fragment order. One wave -> 16x64 strip (4 n-tiles,
// A-fragment reused 4x). 8 waves/block. grid = (N/512, O/16).
// ---------------------------------------------------------------------------
__global__ void __launch_bounds__(256)
wmma_gemm_bias(const unsigned short* __restrict__ A,
               const unsigned short* __restrict__ X,
               const float* __restrict__ bias,
               float* __restrict__ Y,
               int O, int Cp, int N) {
    const int lane  = threadIdx.x & 31;
    const int wave  = threadIdx.x >> 5;
    const int nt0   = (blockIdx.x * 8 + wave) * 4;
    const int otile = blockIdx.y;
    const int Kt    = Cp >> 5;
    const int Nt    = N >> 4;
    const int lb    = lane * 16;

    v8f z = {0.f, 0.f, 0.f, 0.f, 0.f, 0.f, 0.f, 0.f};
    v8f acc[4] = {z, z, z, z};

    const unsigned short* Ap = A + (size_t)otile * Kt * 512 + lb;
    const unsigned short* Xp = X + (size_t)nt0 * 512 + lb;
    const size_t xstep = (size_t)Nt * 512;

    for (int kt = 0; kt < Kt; ++kt) {
        FragBF a;
        a.h[0] = *(const ushort8*)(Ap);
        a.h[1] = *(const ushort8*)(Ap + 8);
        __builtin_prefetch(Ap + 512, 0, 1);      // global_prefetch_b8
        Ap += 512;

        #pragma unroll
        for (int j = 0; j < 4; ++j) {
            FragBF b;
            const unsigned short* xb = Xp + (size_t)j * 512;
            b.h[0] = *(const ushort8*)(xb);
            b.h[1] = *(const ushort8*)(xb + 8);
            acc[j] = __builtin_amdgcn_wmma_f32_16x16x32_bf16(
                         false, a.v, false, b.v, (short)0, acc[j], false, false);
        }
        __builtin_prefetch(Xp + xstep, 0, 1);
        Xp += xstep;
    }

    // C/D layout: VGPR v of lane -> row otile*16 + v + 8*(lane>=16), col n&15
    const int lm = lane & 15, hi = lane >> 4;
    #pragma unroll
    for (int j = 0; j < 4; ++j) {
        int n = (nt0 + j) * 16 + lm;
        #pragma unroll
        for (int v = 0; v < 8; ++v) {
            int row = otile * 16 + v + 8 * hi;
            Y[(size_t)row * N + n] = acc[j][v] + bias[row];
        }
    }
}

// ---------------------------------------------------------------------------
// Weight packing: f32 [O,C] -> bf16 A-fragment tiles [O/16][Cp/32][512].
// A-fragment layout: lane = hi*16 + (o&15); k = (e>=8?16:0) + hi*8 + (e&7).
// ---------------------------------------------------------------------------
__global__ void pack_weight(const float* __restrict__ W,
                            const float* __restrict__ bias,
                            unsigned short* __restrict__ Wb,
                            float* __restrict__ bp,
                            int O, int C, int Cp) {
    int i = blockIdx.x * blockDim.x + threadIdx.x;
    int total = O * Cp;
    if (i < total) {
        int Kt   = Cp >> 5;
        int tile = i >> 9, within = i & 511;
        int lane = within >> 4, e = within & 15;
        int ot = tile / Kt, kt = tile - ot * Kt;
        int hi = lane >> 4;
        int o  = ot * 16 + (lane & 15);
        int k  = kt * 32 + ((e >> 3) << 4) + (hi << 3) + (e & 7);
        Wb[i] = (k < C) ? f2bf(W[(size_t)o * C + k]) : (unsigned short)0;
    }
    if (i < O) bp[i] = bias[i];
}

__global__ void zero_f32k(float* p, int n) {
    int i = blockIdx.x * blockDim.x + threadIdx.x;
    if (i < n) p[i] = 0.f;
}

// Zero padded channel rows [c0, c0+gridDim.y) of a tiled staging buffer.
__global__ void zero_pad_rows(unsigned short* __restrict__ Xb, int c0, int N) {
    int n = blockIdx.x * blockDim.x + threadIdx.x;
    int c = c0 + blockIdx.y;
    if (n >= N) return;
    Xb[bidx(c, n, N)] = 0;
}

// ---------------------------------------------------------------------------
// KNN (exact top-k ascending, stable ties): k passes of lexicographic min.
// xyz [B,P,3]; writes centers [B,M,3] and knn [B*M,K].
// ---------------------------------------------------------------------------
__global__ void knn_select(const float* __restrict__ xyz,
                           float* __restrict__ cxyz,
                           int* __restrict__ knn,
                           int B, int P, int M, int K) {
    int t = blockIdx.x * blockDim.x + threadIdx.x;
    if (t >= B * M) return;
    int b = t / M, m = t - b * M;
    int src = (int)((float)m * (float)(P - 1) / (float)(M - 1));
    const float* xb = xyz + (size_t)b * P * 3;
    float cx = xb[src * 3], cy = xb[src * 3 + 1], cz = xb[src * 3 + 2];
    cxyz[(size_t)t * 3]     = cx;
    cxyz[(size_t)t * 3 + 1] = cy;
    cxyz[(size_t)t * 3 + 2] = cz;
    float lastd = -1.f; int lasti = -1;
    for (int s = 0; s < K; ++s) {
        float bestd = 3.4e38f; int besti = P - 1;
        for (int p = 0; p < P; ++p) {
            float dx = xb[p * 3] - cx, dy = xb[p * 3 + 1] - cy, dz = xb[p * 3 + 2] - cz;
            float d2 = fmaxf(dx * dx + dy * dy + dz * dz, 0.f);
            bool after = (d2 > lastd) || (d2 == lastd && p > lasti);
            if (after && d2 < bestd) { bestd = d2; besti = p; }
        }
        knn[(size_t)t * K + s] = besti;
        lastd = bestd; lasti = besti;
    }
}

// Gather absolute neighbor xyz into tiled bf16 rows 0..2 of staging.
__global__ void gather_xyz(const float* __restrict__ src,
                           const int* __restrict__ knn,
                           unsigned short* __restrict__ Xb,
                           int P, int M, int K, int N) {
    int n = blockIdx.x * blockDim.x + threadIdx.x;
    if (n >= N) return;
    int b = n / (M * K);
    int idx = knn[n];
    const float* q = src + ((size_t)b * P + idx) * 3;
    Xb[bidx(0, n, N)] = f2bf(q[0]);
    Xb[bidx(1, n, N)] = f2bf(q[1]);
    Xb[bidx(2, n, N)] = f2bf(q[2]);
}

// Gather neighbor features: f [Cf, B*Mp] -> tiled rows rowoff.. of staging.
__global__ void gather_feats(const float* __restrict__ f,
                             const int* __restrict__ knn,
                             unsigned short* __restrict__ Xb,
                             int M, int K, int Mp, int N, int rowoff) {
    int n = blockIdx.x * blockDim.x + threadIdx.x;
    int c = blockIdx.y;
    if (n >= N) return;
    int b  = n / (M * K);
    int Bn = N / (M * K);                 // batch count
    int idx = knn[n];
    Xb[bidx(rowoff + c, n, N)] =
        f2bf(f[(size_t)c * Bn * Mp + (size_t)b * Mp + idx]);
}

// ---------------------------------------------------------------------------
// BatchNorm statistics + normalize/ReLU (f32 in place, tiled bf16 out).
// ---------------------------------------------------------------------------
__global__ void bn_stats(const float* __restrict__ Y, float* __restrict__ stats, int N) {
    int c = blockIdx.y;
    __shared__ float ssum[256], ssq[256];
    float s = 0.f, q = 0.f;
    for (size_t i = (size_t)blockIdx.x * blockDim.x + threadIdx.x; i < (size_t)N;
         i += (size_t)gridDim.x * blockDim.x) {
        float v = Y[(size_t)c * N + i];
        s += v; q += v * v;
    }
    ssum[threadIdx.x] = s; ssq[threadIdx.x] = q;
    __syncthreads();
    for (int st = 128; st > 0; st >>= 1) {
        if ((int)threadIdx.x < st) {
            ssum[threadIdx.x] += ssum[threadIdx.x + st];
            ssq[threadIdx.x]  += ssq[threadIdx.x + st];
        }
        __syncthreads();
    }
    if (threadIdx.x == 0) {
        atomicAdd(&stats[2 * c],     ssum[0]);
        atomicAdd(&stats[2 * c + 1], ssq[0]);
    }
}

__global__ void bn_relu(float* __restrict__ Y,
                        const float* __restrict__ stats,
                        const float* __restrict__ g,
                        const float* __restrict__ beta,
                        unsigned short* __restrict__ Xn,
                        int N, size_t total, float invN) {
    size_t i = (size_t)blockIdx.x * blockDim.x + threadIdx.x;
    if (i >= total) return;
    int c = (int)(i / (size_t)N);
    int n = (int)(i - (size_t)c * N);
    float m   = stats[2 * c] * invN;
    float var = stats[2 * c + 1] * invN - m * m;
    float val = (Y[i] - m) * rsqrtf(var + 1e-5f) * g[c] + beta[c];
    val = fmaxf(val, 0.f);
    Y[i] = val;
    Xn[bidx(c, n, N)] = f2bf(val);
}

// Maxpool over the K-sample axis: Y [C, BM*K] -> out [C, BM].
__global__ void maxpool_k(const float* __restrict__ Y, float* __restrict__ out,
                          int C, int BM, int K) {
    int t = blockIdx.x * blockDim.x + threadIdx.x;
    if (t >= C * BM) return;
    int c = t / BM, j = t - c * BM;
    const float* row = Y + (size_t)c * BM * K + (size_t)j * K;
    float m = row[0];
    for (int s = 1; s < K; ++s) m = fmaxf(m, row[s]);
    out[t] = m;
}

// ---------------------------------------------------------------------------
// Three-NN inverse-distance interpolation prep: per point, 3 nearest + weights.
// ---------------------------------------------------------------------------
__global__ void knn3_prep(const float* __restrict__ xyz1,
                          const float* __restrict__ xyz2,
                          int* __restrict__ idx3, float* __restrict__ w3,
                          int B, int P1, int P2) {
    int t = blockIdx.x * blockDim.x + threadIdx.x;
    if (t >= B * P1) return;
    int b = t / P1;
    const float* q = xyz1 + (size_t)t * 3;
    float cx = q[0], cy = q[1], cz = q[2];
    const float* xb = xyz2 + (size_t)b * P2 * 3;
    float lastd = -1.f; int lasti = -1;
    float dd[3]; int ii[3];
    for (int j = 0; j < 3; ++j) {
        float bestd = 3.4e38f; int besti = P2 - 1;
        for (int p = 0; p < P2; ++p) {
            float dx = xb[p * 3] - cx, dy = xb[p * 3 + 1] - cy, dz = xb[p * 3 + 2] - cz;
            float d2 = fmaxf(dx * dx + dy * dy + dz * dz, 0.f);
            bool after = (d2 > lastd) || (d2 == lastd && p > lasti);
            if (after && d2 < bestd) { bestd = d2; besti = p; }
        }
        dd[j] = fmaxf(sqrtf(bestd), 1e-8f);
        ii[j] = besti;
        lastd = bestd; lasti = besti;
    }
    float w0 = 1.f / dd[0], w1 = 1.f / dd[1], w2 = 1.f / dd[2];
    float s = w0 + w1 + w2;
    w3[(size_t)t * 3]     = w0 / s;
    w3[(size_t)t * 3 + 1] = w1 / s;
    w3[(size_t)t * 3 + 2] = w2 / s;
    idx3[(size_t)t * 3]     = ii[0];
    idx3[(size_t)t * 3 + 1] = ii[1];
    idx3[(size_t)t * 3 + 2] = ii[2];
}

// interp: tiled row c of staging = sum_j w_j * f2[c, b*P2 + idx_j]
__global__ void interp_write(const float* __restrict__ f2,
                             const int* __restrict__ idx3,
                             const float* __restrict__ w3,
                             unsigned short* __restrict__ Xb,
                             int P1, int P2, int N, int ldf) {
    int n = blockIdx.x * blockDim.x + threadIdx.x;
    int c = blockIdx.y;
    if (n >= N) return;
    int b = n / P1;
    const float* fc = f2 + (size_t)c * ldf + (size_t)b * P2;
    float acc = w3[(size_t)n * 3]     * fc[idx3[(size_t)n * 3]]
              + w3[(size_t)n * 3 + 1] * fc[idx3[(size_t)n * 3 + 1]]
              + w3[(size_t)n * 3 + 2] * fc[idx3[(size_t)n * 3 + 2]];
    Xb[bidx(c, n, N)] = f2bf(acc);
}

// f32 [C1,N] -> tiled bf16 rows rowoff.. of staging (skip-connection concat).
__global__ void copy_rows_f2bf(const float* __restrict__ src,
                               unsigned short* __restrict__ Xb,
                               int rowoff, int N) {
    int n = blockIdx.x * blockDim.x + threadIdx.x;
    int c = blockIdx.y;
    if (n >= N) return;
    Xb[bidx(rowoff + c, n, N)] = f2bf(src[(size_t)c * N + n]);
}

// Final linear head: out[n,o] = b2[o] + sum_c W2[o,c] * h[c,n], O=10, C=128.
__global__ void head2_kernel(const float* __restrict__ h,
                             const float* __restrict__ W2,
                             const float* __restrict__ b2,
                             float* __restrict__ out, int N) {
    int n = blockIdx.x * blockDim.x + threadIdx.x;
    if (n >= N) return;
    for (int o = 0; o < 10; ++o) {
        float acc = b2[o];
        for (int c = 0; c < 128; ++c)
            acc += W2[o * 128 + c] * h[(size_t)c * N + n];
        out[(size_t)n * 10 + o] = acc;
    }
}

// ---------------------------------------------------------------------------
// Host orchestration
// ---------------------------------------------------------------------------
static inline int cp32(int c) { return (c + 31) & ~31; }

extern "C" void kernel_launch(void* const* d_in, const int* in_sizes, int n_in,
                              void* d_out, int out_size, void* d_ws, size_t ws_size,
                              hipStream_t stream) {
    (void)in_sizes; (void)n_in; (void)out_size; (void)ws_size;

    const int B = 8, P0 = 8192;
    const int M1 = 2048, K1 = 32, M2 = 512, K2 = 64, M3 = 128, K3 = 128;
    const int N1 = B * M1 * K1;   // 524288
    const int N2 = B * M2 * K2;   // 262144
    const int N3 = B * M3 * K3;   // 131072
    const int NP1 = B * M1, NP2 = B * M2, NP3 = B * M3, NP0 = B * P0;

    // ---- input pointer walk (setup_inputs order) ----
    int pi = 0;
    const float* xyz = (const float*)d_in[pi++];
    struct Layer {
        const float *W, *b, *g, *beta;
        int O, C, Cp;
        unsigned short* Wb;
        float* bp;
    };
    Layer Ls[17];
    int li = 0;
    auto addL = [&](int O, int C) {
        Layer& L = Ls[li++];
        L.W    = (const float*)d_in[pi++];
        L.b    = (const float*)d_in[pi++];
        L.g    = (const float*)d_in[pi++];
        L.beta = (const float*)d_in[pi++];
        L.O = O; L.C = C; L.Cp = cp32(C);
    };
    addL(64, 3);    addL(64, 64);    addL(128, 64);        // sa1
    addL(128, 131); addL(128, 128);  addL(256, 128);       // sa2
    addL(256, 259); addL(512, 256);  addL(1024, 512);      // sa3
    addL(256, 1280); addL(256, 256);                        // fp3
    addL(256, 384);  addL(128, 256);                        // fp2
    addL(128, 128);  addL(128, 128); addL(128, 128);       // fp1
    addL(128, 128);                                         // head1
    const float* W2h = (const float*)d_in[pi++];
    const float* b2h = (const float*)d_in[pi++];

    // ---- workspace bump allocator ----
    char* base = (char*)d_ws;
    size_t off = 0;
    auto alloc = [&](size_t bytes) -> void* {
        void* r = base + off;
        off = (off + bytes + 255) & ~(size_t)255;
        return r;
    };
    for (int i = 0; i < 17; ++i) {
        Ls[i].Wb = (unsigned short*)alloc((size_t)Ls[i].O * Ls[i].Cp * 2);
        Ls[i].bp = (float*)alloc((size_t)Ls[i].O * 4);
    }
    const size_t SLOT = (size_t)1024 * N3;                 // 134M elems
    unsigned short* slotA = (unsigned short*)alloc(SLOT * 2);
    unsigned short* slotB = (unsigned short*)alloc(SLOT * 2);
    float* Yf = (float*)alloc(SLOT * 4);
    float* c1 = (float*)alloc((size_t)NP1 * 3 * 4);
    float* c2 = (float*)alloc((size_t)NP2 * 3 * 4);
    float* c3 = (float*)alloc((size_t)NP3 * 3 * 4);
    int*   knn  = (int*)alloc((size_t)NP1 * K1 * 4);
    float* l1   = (float*)alloc((size_t)128 * NP1 * 4);
    float* l2   = (float*)alloc((size_t)256 * NP2 * 4);
    float* l3   = (float*)alloc((size_t)1024 * NP3 * 4);
    int*   idx3 = (int*)alloc((size_t)NP0 * 3 * 4);
    float* w3   = (float*)alloc((size_t)NP0 * 3 * 4);
    float* stats = (float*)alloc(2048 * 4);

    // ---- pack weights to bf16 fragment tiles ----
    for (int i = 0; i < 17; ++i) {
        int total = Ls[i].O * Ls[i].Cp;
        pack_weight<<<(total + 255) / 256, 256, 0, stream>>>(
            Ls[i].W, Ls[i].b, Ls[i].Wb, Ls[i].bp, Ls[i].O, Ls[i].C, Ls[i].Cp);
    }

    auto pad_rows = [&](unsigned short* Xb, int C, int Cp, int N) {
        if (Cp > C) {
            dim3 g((N + 255) / 256, Cp - C);
            zero_pad_rows<<<g, 256, 0, stream>>>(Xb, C, N);
        }
    };
    auto gemm = [&](const Layer& L, const unsigned short* X, int N) {
        dim3 g(N / 512, L.O / 16);
        wmma_gemm_bias<<<g, 256, 0, stream>>>(L.Wb, X, L.bp, Yf, L.O, L.Cp, N);
    };
    auto bn = [&](const Layer& L, unsigned short* Xn, int N) {
        zero_f32k<<<(2 * L.O + 255) / 256, 256, 0, stream>>>(stats, 2 * L.O);
        dim3 g(64, L.O);
        bn_stats<<<g, 256, 0, stream>>>(Yf, stats, N);
        size_t total = (size_t)L.O * N;
        bn_relu<<<(unsigned)((total + 255) / 256), 256, 0, stream>>>(
            Yf, stats, L.g, L.beta, Xn, N, total, 1.0f / (float)N);
    };
    // Runs layers [first, first+nL) starting from tiled bf16 input `in`.
    // Returns the slot holding the final bf16 activations; final f32 is in Yf.
    auto run_mlp = [&](int first, int nL, int N, unsigned short* in) -> unsigned short* {
        unsigned short* cur = in;
        unsigned short* oth = (in == slotA) ? slotB : slotA;
        for (int l = 0; l < nL; ++l) {
            const Layer& L = Ls[first + l];
            gemm(L, cur, N);
            bn(L, oth, N);
            unsigned short* t = cur; cur = oth; oth = t;
        }
        return cur;
    };

    // ================= SA1 =================
    knn_select<<<(NP1 + 127) / 128, 128, 0, stream>>>(xyz, c1, knn, B, P0, M1, K1);
    pad_rows(slotA, 3, 32, N1);
    gather_xyz<<<(N1 + 255) / 256, 256, 0, stream>>>(xyz, knn, slotA, P0, M1, K1, N1);
    run_mlp(0, 3, N1, slotA);
    maxpool_k<<<(128 * NP1 + 255) / 256, 256, 0, stream>>>(Yf, l1, 128, NP1, K1);

    // ================= SA2 =================
    knn_select<<<(NP2 + 127) / 128, 128, 0, stream>>>(c1, c2, knn, B, M1, M2, K2);
    pad_rows(slotA, 131, 160, N2);
    gather_xyz<<<(N2 + 255) / 256, 256, 0, stream>>>(c1, knn, slotA, M1, M2, K2, N2);
    {
        dim3 g((N2 + 255) / 256, 128);
        gather_feats<<<g, 256, 0, stream>>>(l1, knn, slotA, M2, K2, M1, N2, 3);
    }
    run_mlp(3, 3, N2, slotA);
    maxpool_k<<<(256 * NP2 + 255) / 256, 256, 0, stream>>>(Yf, l2, 256, NP2, K2);

    // ================= SA3 =================
    knn_select<<<(NP3 + 127) / 128, 128, 0, stream>>>(c2, c3, knn, B, M2, M3, K3);
    pad_rows(slotA, 259, 288, N3);
    gather_xyz<<<(N3 + 255) / 256, 256, 0, stream>>>(c2, knn, slotA, M2, M3, K3, N3);
    {
        dim3 g((N3 + 255) / 256, 256);
        gather_feats<<<g, 256, 0, stream>>>(l2, knn, slotA, M3, K3, M2, N3, 3);
    }
    run_mlp(6, 3, N3, slotA);
    maxpool_k<<<(1024 * NP3 + 255) / 256, 256, 0, stream>>>(Yf, l3, 1024, NP3, K3);

    // ================= FP3: l3 -> l2 level =================
    knn3_prep<<<(NP2 + 127) / 128, 128, 0, stream>>>(c2, c3, idx3, w3, B, M2, M3);
    {
        dim3 g((NP2 + 255) / 256, 1024);
        interp_write<<<g, 256, 0, stream>>>(l3, idx3, w3, slotA, M2, M3, NP2, NP3);
    }
    {
        dim3 g((NP2 + 255) / 256, 256);
        copy_rows_f2bf<<<g, 256, 0, stream>>>(l2, slotA, 1024, NP2);
    }
    run_mlp(9, 2, NP2, slotA);            // Yf = l2n [256, NP2]

    // ================= FP2: l2n -> l1 level =================
    knn3_prep<<<(NP1 + 127) / 128, 128, 0, stream>>>(c1, c2, idx3, w3, B, M1, M2);
    {
        dim3 g((NP1 + 255) / 256, 256);
        interp_write<<<g, 256, 0, stream>>>(Yf, idx3, w3, slotA, M1, M2, NP1, NP2);
    }
    {
        dim3 g((NP1 + 255) / 256, 128);
        copy_rows_f2bf<<<g, 256, 0, stream>>>(l1, slotA, 256, NP1);
    }
    run_mlp(11, 2, NP1, slotA);           // Yf = l1n [128, NP1]

    // ================= FP1: l1n -> full resolution =================
    knn3_prep<<<(NP0 + 127) / 128, 128, 0, stream>>>(xyz, c1, idx3, w3, B, P0, M1);
    {
        dim3 g((NP0 + 255) / 256, 128);
        interp_write<<<g, 256, 0, stream>>>(Yf, idx3, w3, slotA, P0, M1, NP0, NP1);
    }
    unsigned short* fb = run_mlp(13, 3, NP0, slotA);   // Yf = l0n [128, NP0]

    // ================= head =================
    run_mlp(16, 1, NP0, fb);              // Yf = h [128, NP0]
    head2_kernel<<<(NP0 + 255) / 256, 256, 0, stream>>>(Yf, W2h, b2h,
                                                        (float*)d_out, NP0);
}
